// LSTMClassifier_55997783605841
// MI455X (gfx1250) — compile-verified, matrix operand order checked by reading, not run on hardware
//
#include <hip/hip_runtime.h>

// ---------------------------------------------------------------------------
// Fully-fused 2-layer LSTM classifier for MI455X (gfx1250, wave32, WMMA).
//
// B=1024, T=512, V=8, E=32, H=64, gates=4H=256.
//
// v3 (final polish over v2):
//  * layer-1 bias hoisted to registers (was 4 LDS loads/step).
//  * x-index load software-pipelined one step ahead (latency hidden behind
//    the whole step body + global_prefetch_b8).
// Design recap:
//  * gate(N)-dim split 4-ways across waves per 16-row batch tile: 28 v_wmma
//    per wave per step; 32 blocks x 8 waves = 256 waves (2/SIMD).
//  * layer-0 input GEMM folded into the recurrence via one-hot K-augmentation
//    (V=8 -> 8x256 table G0 as extra K rows); layer-1 input+recurrent GEMMs
//    merged into one K=128 GEMM on [h0(t) | h1(t-1)].
//  * all weights LDS-resident, pre-swizzled into exact WMMA B-operand
//    per-lane streams; h-state ping-pongs through LDS in A-operand layout;
//    c-state lives in accumulator-layout VGPRs.  No HBM intermediates:
//    traffic = 2 MB indices + 115 KB weights + 32 KB logits.
// ---------------------------------------------------------------------------

#define B_SZ  1024
#define T_SZ  512
#define V_SZ  8
#define E_SZ  32
#define H_SZ  64
#define NG    256                 // 4*H gates

typedef __attribute__((ext_vector_type(16))) __bf16       v16bf;
typedef __attribute__((ext_vector_type(8)))  float        v8f;
typedef __attribute__((ext_vector_type(8)))  unsigned int v8u;
typedef __attribute__((ext_vector_type(4)))  unsigned int u32x4;

// ws / LDS byte layout ------------------------------------------------------
// W0 stream: [ks=3][nt=16][lane=32][e=16] bf16   (K=96 aug: Whh0^T ; G0 ; 0)
// W1 stream: [ks=4][nt=16][lane=32][e=16] bf16   (K=128: Wih1^T ; Whh1^T)
#define WS_W0   0
#define WS_W1   49152            // 3*16*32*16*2
#define WS_B1   114688           // + 4*16*32*16*2
#define WS_TOT  115712           // + 256*4
// per-tile h scratch: [h0 ping|h0 pong|h1 ping|h1 pong] x 1024 u16 = 8192 B
#define LDS_H   115712
#define LDS_TOT (115712 + 2 * 8192)

static __device__ __forceinline__ unsigned short f2bf(float f) {
  unsigned u = __builtin_bit_cast(unsigned, f);
  unsigned r = u + 0x7FFFu + ((u >> 16) & 1u);     // round-to-nearest-even
  return (unsigned short)(r >> 16);
}
static __device__ __forceinline__ float sigm(float x) {
  return 1.0f / (1.0f + __expf(-x));
}
static __device__ __forceinline__ float tanh_(float x) {
  float e = __expf(2.0f * x);
  return (e - 1.0f) / (e + 1.0f);
}
static __device__ __forceinline__ v8f wmma_bf16(v16bf a, v16bf b, v8f c) {
  return __builtin_amdgcn_wmma_f32_16x16x32_bf16(false, a, false, b, (short)0, c,
                                                 false, false);
}
// 16-bit A/B per-lane element e -> local K  (matches ISA 16-bit A 16x32 table)
static __device__ __forceinline__ int klocal(int e, int half) {
  return (e < 8 ? 0 : 16) + half * 8 + (e & 7);
}
static __device__ __forceinline__ v16bf ld_op32B(const unsigned short* base, int idx) {
  // idx in units of 16 u16 elements (32 B), 32B-aligned -> ds_load_b128 x2
  const v8u* p = (const v8u*)(base + idx * 16);
  return __builtin_bit_cast(v16bf, *p);
}

// ---------------------------------------------------------------------------
// Prep kernel: build G0 = emb @ Wih0^T + bih0 + bhh0, then swizzle W0/W1 into
// WMMA B-operand streams (bf16) and bias1 into ws.  One 256-thread block.
// ---------------------------------------------------------------------------
__global__ __launch_bounds__(256) void lstm_prep(
    const float* __restrict__ emb,  const float* __restrict__ Wih0,
    const float* __restrict__ Whh0, const float* __restrict__ bih0,
    const float* __restrict__ bhh0, const float* __restrict__ Wih1,
    const float* __restrict__ Whh1, const float* __restrict__ bih1,
    const float* __restrict__ bhh1, char* __restrict__ ws) {
  __shared__ float G0s[V_SZ][NG];
  const int n = threadIdx.x;                       // 0..255 : gate index
  for (int v = 0; v < V_SZ; ++v) {
    float s = bih0[n] + bhh0[n];
    #pragma unroll
    for (int e = 0; e < E_SZ; ++e) s += emb[v * E_SZ + e] * Wih0[n * E_SZ + e];
    G0s[v][n] = s;
  }
  __syncthreads();

  unsigned short* W0 = (unsigned short*)(ws + WS_W0);
  unsigned short* W1 = (unsigned short*)(ws + WS_W1);
  float*          B1 = (float*)(ws + WS_B1);

  // W0 stream: K-aug 96 = [Whh0^T (64) ; G0 (8) ; zero (24)]
  for (int i = n; i < 3 * 16 * 32 * 16; i += 256) {
    int e = i & 15, lane = (i >> 4) & 31, nt = (i >> 9) & 15, ks = i >> 13;
    int Kg = ks * 32 + klocal(e, (lane >> 4) & 1);
    int nn = nt * 16 + (lane & 15);
    float val = 0.0f;
    if (Kg < 64)      val = Whh0[nn * H_SZ + Kg];       // Whh0^T[k][n]
    else if (Kg < 72) val = G0s[Kg - 64][nn];           // one-hot rows
    W0[i] = f2bf(val);
  }
  // W1 stream: K=128 = [Wih1^T (64) ; Whh1^T (64)]
  for (int i = n; i < 4 * 16 * 32 * 16; i += 256) {
    int e = i & 15, lane = (i >> 4) & 31, nt = (i >> 9) & 15, ks = i >> 13;
    int Kg = ks * 32 + klocal(e, (lane >> 4) & 1);
    int nn = nt * 16 + (lane & 15);
    float val = (Kg < 64) ? Wih1[nn * H_SZ + Kg] : Whh1[nn * H_SZ + (Kg - 64)];
    W1[i] = f2bf(val);
  }
  B1[n] = bih1[n] + bhh1[n];
}

// ---------------------------------------------------------------------------
// Main kernel: 32 blocks x 256 threads (8 waves = 2 batch tiles x 4 waves).
// Wave (tile, g): batch rows [ (blk*2+tile)*16, +16 ), gate column-group g
// (h columns g*16 .. g*16+15; gate tiles {g, 4+g, 8+g, 12+g}).
// ---------------------------------------------------------------------------
__global__ __launch_bounds__(256) void lstm_main(
    const int* __restrict__ xp, const float* __restrict__ Wfc,
    const float* __restrict__ bfc, const char* __restrict__ ws,
    float* __restrict__ out) {
  extern __shared__ char smem[];
  const int tid  = threadIdx.x;
  const int lane = tid & 31;
  const int wave = tid >> 5;           // 0..7
  const int tile = wave >> 2;          // 0..1 : batch tile within block
  const int g    = wave & 3;           // gate column-group this wave owns
  const int half = lane >> 4;          // lane half-select (K group / M group)
  const int nloc = lane & 15;          // N (column) within a 16-tile
  const int r0   = (blockIdx.x * 2 + tile) * 16;

  // Stage weights+bias ws -> LDS (115712 B = 7232 x 16B).
  {
    const u32x4* src = (const u32x4*)ws;
    u32x4*       dst = (u32x4*)smem;
    for (int i = tid; i < WS_TOT / 16; i += 256) dst[i] = src[i];
  }
  // zero both tiles' h ping-pong buffers (8192 u16 total)
  {
    unsigned short* hz = (unsigned short*)(smem + LDS_H);
    for (int i = tid; i < 2 * 4096; i += 256) hz[i] = 0;
  }
  __syncthreads();

  const unsigned short* W0  = (const unsigned short*)(smem + WS_W0);
  const unsigned short* W1  = (const unsigned short*)(smem + WS_W1);
  const float*          B1s = (const float*)(smem + WS_B1);
  unsigned short* tb = (unsigned short*)(smem + LDS_H + tile * 8192);
  // tb: [0..1023]=h0 buf0, [1024..2047]=h0 buf1, [2048..]=h1 buf0/1

  // loop-invariant layer-1 bias for this wave's 4 gate tiles -> registers
  float b1r[4];
  #pragma unroll
  for (int gg = 0; gg < 4; ++gg) b1r[gg] = B1s[(gg * 4 + g) * 16 + nloc];

  // c-state for this wave's column group only (C layout, 8 VGPRs each)
  float c0[8], c1[8];
  #pragma unroll
  for (int v = 0; v < 8; ++v) { c0[v] = 0.0f; c1[v] = 0.0f; }

  // h-store address pieces for this wave's columns (constant per lane)
  const int col = g * 16 + nloc;
  const int aopS = col >> 5, KlS = col & 31;
  const int h2S = (KlS >> 3) & 1, eS = ((KlS >> 4) & 1) * 8 + (KlS & 7);
  const int stBase = (aopS * 32 + h2S * 16 + half * 8) * 16 + eS; // + v*16

  // x-index software pipeline: idx for step t loaded during step t-1
  const int* xrow = xp + (r0 + nloc) * T_SZ;
  int idx_cur = xrow[0];

  for (int t = 0; t < T_SZ; ++t) {
    const int p = t & 1;               // read buffer
    unsigned short* h0p = tb + p * 1024;
    unsigned short* h0w = tb + (1 - p) * 1024;
    unsigned short* h1p = tb + 2048 + p * 1024;
    unsigned short* h1w = tb + 2048 + (1 - p) * 1024;

    // issue next step's index load + prefetch right away (hidden by body)
    int idx_next = 0;
    if (t + 1 < T_SZ) idx_next = xrow[t + 1];
    if (t + 16 < T_SZ) __builtin_prefetch(&xrow[t + 16], 0, 1);

    // ---- per-step one-hot A operand (K=64..95 of layer-0 aug GEMM) -------
    // bf16 1.0 sits in pair q=idx>>1, lo/hi=idx&1, only in lane-half 0.
    const int      ptarget = (half == 0) ? (idx_cur >> 1) : 99;
    const unsigned pword   = (idx_cur & 1) ? 0x3F800000u : 0x00003F80u;
    v8u a2u;
    #pragma unroll
    for (int q = 0; q < 8; ++q) a2u[q] = (q == ptarget) ? pword : 0u;

    // ---- layer 0: this wave's 4 gate tiles {g,4+g,8+g,12+g}, K=96 --------
    v16bf Aarr[3];
    Aarr[0] = ld_op32B(h0p, lane);
    Aarr[1] = ld_op32B(h0p, 32 + lane);
    Aarr[2] = __builtin_bit_cast(v16bf, a2u);
    {
      v8f acc[4];
      #pragma unroll
      for (int gg = 0; gg < 4; ++gg) acc[gg] = (v8f){};
      v16bf bcur = ld_op32B(W0, (0 * 16 + g) * 32 + lane);
      #pragma unroll
      for (int j = 0; j < 12; ++j) {               // j = gg*3 + ks
        const int gg = j / 3, ks = j % 3;
        v16bf bnext = bcur;
        if (j < 11) {
          const int jn = j + 1, ggn = jn / 3, ksn = jn % 3;
          bnext = ld_op32B(W0, (ksn * 16 + (ggn * 4 + g)) * 32 + lane);
        }
        acc[gg] = wmma_bf16(Aarr[ks], bcur, acc[gg]);
        bcur = bnext;
      }
      #pragma unroll
      for (int v = 0; v < 8; ++v) {
        const float iv = sigm(acc[0][v]), fv = sigm(acc[1][v]);
        const float gv = tanh_(acc[2][v]), ov = sigm(acc[3][v]);
        const float c = fv * c0[v] + iv * gv;
        c0[v] = c;
        h0w[stBase + v * 16] = f2bf(ov * tanh_(c));  // A-layout store
      }
    }
    __syncthreads();   // h0(t) complete across the 4 waves of this tile

    // ---- layer 1: gates = [h0(t) | h1(t-1)] @ [Wih1^T ; Whh1^T], K=128 ---
    v16bf Barr[4];
    Barr[0] = ld_op32B(h0w, lane);
    Barr[1] = ld_op32B(h0w, 32 + lane);
    Barr[2] = ld_op32B(h1p, lane);
    Barr[3] = ld_op32B(h1p, 32 + lane);
    {
      v8f acc[4];
      #pragma unroll
      for (int gg = 0; gg < 4; ++gg) {
        const float b1 = b1r[gg];
        acc[gg] = (v8f){b1, b1, b1, b1, b1, b1, b1, b1};
      }
      v16bf bcur = ld_op32B(W1, (0 * 16 + g) * 32 + lane);
      #pragma unroll
      for (int j = 0; j < 16; ++j) {               // j = gg*4 + ks
        const int gg = j / 4, ks = j % 4;
        v16bf bnext = bcur;
        if (j < 15) {
          const int jn = j + 1, ggn = jn / 4, ksn = jn % 4;
          bnext = ld_op32B(W1, (ksn * 16 + (ggn * 4 + g)) * 32 + lane);
        }
        acc[gg] = wmma_bf16(Barr[ks], bcur, acc[gg]);
        bcur = bnext;
      }
      #pragma unroll
      for (int v = 0; v < 8; ++v) {
        const float iv = sigm(acc[0][v]), fv = sigm(acc[1][v]);
        const float gv = tanh_(acc[2][v]), ov = sigm(acc[3][v]);
        const float c = fv * c1[v] + iv * gv;
        c1[v] = c;
        h1w[stBase + v * 16] = f2bf(ov * tanh_(c));
      }
    }
    __syncthreads();   // h1(t) complete before next step reads it

    idx_cur = idx_next;
  }

  // ---- classifier head: logits[16,8] = h1_last @ Wfc^T + bfc -------------
  // last write went to buffer 1-(511&1) = 0  ->  tb+2048
  const unsigned short* h1f = tb + 2048;
  const int m  = nloc;                       // row this lane handles
  const int vo = g * 2 + half;               // output class 0..7
  float s = bfc[vo];
  for (int k = 0; k < H_SZ; ++k) {
    const int aop = k >> 5, Kl = k & 31;
    const int h2 = (Kl >> 3) & 1, e = ((Kl >> 4) & 1) * 8 + (Kl & 7);
    const unsigned short raw = h1f[(aop * 32 + h2 * 16 + m) * 16 + e];
    s += __builtin_bit_cast(float, (unsigned)raw << 16) * Wfc[vo * H_SZ + k];
  }
  out[(r0 + m) * V_SZ + vo] = s;
}

// ---------------------------------------------------------------------------
extern "C" void kernel_launch(void* const* d_in, const int* in_sizes, int n_in,
                              void* d_out, int out_size, void* d_ws, size_t ws_size,
                              hipStream_t stream) {
  const int*   x    = (const int*)d_in[0];
  const float* emb  = (const float*)d_in[1];
  const float* Wih0 = (const float*)d_in[2];
  const float* Whh0 = (const float*)d_in[3];
  const float* bih0 = (const float*)d_in[4];
  const float* bhh0 = (const float*)d_in[5];
  const float* Wih1 = (const float*)d_in[6];
  const float* Whh1 = (const float*)d_in[7];
  const float* bih1 = (const float*)d_in[8];
  const float* bhh1 = (const float*)d_in[9];
  const float* Wfc  = (const float*)d_in[10];
  const float* bfc  = (const float*)d_in[11];
  float* out = (float*)d_out;
  char*  ws  = (char*)d_ws;

  // Opt in to >64KB dynamic LDS (CDNA5 WGP has 320KB).
  (void)hipFuncSetAttribute((const void*)lstm_main,
                            hipFuncAttributeMaxDynamicSharedMemorySize, LDS_TOT);

  lstm_prep<<<dim3(1), dim3(256), 0, stream>>>(emb, Wih0, Whh0, bih0, bhh0,
                                               Wih1, Whh1, bih1, bhh1, ws);
  lstm_main<<<dim3(32), dim3(256), LDS_TOT, stream>>>(x, Wfc, bfc, ws, out);
}
